// Qwen3DFlashAttention_36395552866945
// MI455X (gfx1250) — compile-verified, matrix operand order checked by reading, not run on hardware
//
#include <hip/hip_runtime.h>

typedef __attribute__((ext_vector_type(16))) _Float16 v16h;
typedef __attribute__((ext_vector_type(8)))  _Float16 v8h;
typedef __attribute__((ext_vector_type(8)))  float    v8f;

static __device__ __forceinline__ v8f zero8() {
    v8f z = {0.f, 0.f, 0.f, 0.f, 0.f, 0.f, 0.f, 0.f};
    return z;
}

// Load a 16x32 f16 fragment (A layout per CDNA5 ISA; B uses the same layout
// with the matrix stored column-major, i.e. base indexes [n][k]).
// lane<16:  row = lane,    K = {kb..kb+7, kb+16..kb+23}, kb=0
// lane>=16: row = lane-16, kb=8
static __device__ __forceinline__ v16h load_frag(const _Float16* p, int ld) {
    const int lane = threadIdx.x & 31;
    const int r    = lane & 15;
    const int kb   = (lane < 16) ? 0 : 8;
    const _Float16* q = p + (size_t)r * ld + kb;
    v8h lo = *(const v8h*)(q);
    v8h hi = *(const v8h*)(q + 16);
    v16h f;
#pragma unroll
    for (int i = 0; i < 8; ++i) { f[i] = lo[i]; f[i + 8] = hi[i]; }
    return f;
}

static __device__ __forceinline__ v8f wmma32(v16h a, v16h b, v8f c) {
    return __builtin_amdgcn_wmma_f32_16x16x32_f16(false, a, false, b,
                                                  (short)0, c, false, false);
}

// Async copy of 64 contiguous bytes global->LDS using the CDNA5 TDM-class
// GLOBAL_LOAD_ASYNC_TO_LDS_B128 path (ASYNCcnt). INST_OFFSET is added to both
// the LDS and the global address, so one base pair serves all four beats.
static __device__ __forceinline__ void async_copy64(unsigned lds_addr,
                                                    const void* gaddr) {
    unsigned long long ga = (unsigned long long)gaddr;
    asm volatile(
        "global_load_async_to_lds_b128 %0, %1, off\n\t"
        "global_load_async_to_lds_b128 %0, %1, off offset:16\n\t"
        "global_load_async_to_lds_b128 %0, %1, off offset:32\n\t"
        "global_load_async_to_lds_b128 %0, %1, off offset:48"
        :: "v"(lds_addr), "v"(ga) : "memory");
}

static __device__ __forceinline__ void wait_async0() {
    asm volatile("s_wait_asynccnt 0x0" ::: "memory");
}

// ---------------------------------------------------------------- cast f32->f16
__global__ __launch_bounds__(256) void cast_f16_kernel(const float* __restrict__ s,
                                                       _Float16* __restrict__ d, int n) {
    int i = blockIdx.x * 256 + threadIdx.x;
    if (i < n) d[i] = (_Float16)s[i];
}

// ---------------------------------------------------------------- WMMA GEMM
// C(MxN,f32) = A(MxK,f16,row) * B(KxN,f16,row). Block tile 128x64, BK=64.
// 8 waves -> 4x2 wave grid, each wave 32x32 = 2x2 WMMA tiles.
// A tile staged with async global->LDS (no VGPR round-trip); B tile transposed
// through VGPRs into Bs[n][k] so both fragment loads are ds_load_b128.
__global__ __launch_bounds__(256) void gemm_wmma(const _Float16* __restrict__ A,
                                                 const _Float16* __restrict__ B,
                                                 float* __restrict__ C,
                                                 int M, int N, int K) {
    __shared__ _Float16 As[128 * 72];
    __shared__ _Float16 Bs[64 * 72];
    const int tid  = threadIdx.x;
    const int wid  = tid >> 5;
    const int lane = tid & 31;
    const int wm = wid >> 1, wn = wid & 1;
    const int m0 = blockIdx.y * 128, n0 = blockIdx.x * 64;
    const int nl = lane & 15;
    const int mb = (lane >> 4) << 3;

    // per-thread staging coordinates (constant over the K loop)
    const int arow = tid >> 1, akh = (tid & 1) * 32;
    const unsigned alds = (unsigned)(size_t)(void*)&As[arow * 72 + akh];
    const _Float16* agp = A + (size_t)(m0 + arow) * K + akh;
    const int bk = tid >> 2, bn0 = (tid & 3) * 16;
    const _Float16* bgp = B + (size_t)bk * N + n0 + bn0;

    v8f acc[2][2];
#pragma unroll
    for (int a = 0; a < 2; ++a)
#pragma unroll
        for (int b = 0; b < 2; ++b) acc[a][b] = zero8();

    for (int k0 = 0; k0 < K; k0 += 64) {
        __syncthreads();
        // A: 128 rows x 64 halves; each thread copies 64B via async-to-LDS
        async_copy64(alds, agp + k0);
        {   // B: transpose 64x64 tile into Bs[n][k]
            const _Float16* src = bgp + (size_t)k0 * N;
            v8h b0 = *(const v8h*)src;
            v8h b1 = *(const v8h*)(src + 8);
#pragma unroll
            for (int i = 0; i < 8; ++i) {
                Bs[(bn0 + i) * 72 + bk]     = b0[i];
                Bs[(bn0 + 8 + i) * 72 + bk] = b1[i];
            }
        }
        wait_async0();
        __syncthreads();
#pragma unroll
        for (int s = 0; s < 2; ++s) {
            v16h af[2], bf[2];
#pragma unroll
            for (int t = 0; t < 2; ++t)
                af[t] = load_frag(&As[(wm * 32 + t * 16) * 72 + s * 32], 72);
#pragma unroll
            for (int t = 0; t < 2; ++t)
                bf[t] = load_frag(&Bs[(wn * 32 + t * 16) * 72 + s * 32], 72);
#pragma unroll
            for (int tm = 0; tm < 2; ++tm)
#pragma unroll
                for (int tn = 0; tn < 2; ++tn)
                    acc[tm][tn] = wmma32(af[tm], bf[tn], acc[tm][tn]);
        }
    }
    // epilogue: one base pointer, constant-strided stores
    float* cp = C + (size_t)(m0 + wm * 32 + mb) * N + (n0 + wn * 32 + nl);
#pragma unroll
    for (int tm = 0; tm < 2; ++tm)
#pragma unroll
        for (int r = 0; r < 8; ++r) {
            float* rp = cp + (size_t)(tm * 16 + r) * N;
            rp[0]  = acc[tm][0][r];
            rp[16] = acc[tm][1][r];
        }
}

// ---------------------------------------------------------------- RMSNorm + RoPE
// src: rows x (heads*64) f32. dst: [h][rows][64] f16. One wave32 per (row,h).
__global__ __launch_bounds__(256) void norm_rope(const float* __restrict__ src,
                                                 const float* __restrict__ cosp,
                                                 const float* __restrict__ sinp,
                                                 const float* __restrict__ w,
                                                 _Float16* __restrict__ dst,
                                                 int rows, int heads, int ld, int posoff) {
    const int wslot = blockIdx.x * 8 + (threadIdx.x >> 5);
    const int lane  = threadIdx.x & 31;
    if (wslot >= rows * heads) return;
    const int row = wslot / heads;
    const int h   = wslot - row * heads;
    const float* x = src + (size_t)row * ld + h * 64;
    float x0 = x[lane], x1 = x[lane + 32];
    float ss = x0 * x0 + x1 * x1;
#pragma unroll
    for (int o = 16; o >= 1; o >>= 1) ss += __shfl_xor(ss, o, 32);
    const float sc = rsqrtf(ss * (1.0f / 64.0f) + 1e-6f);
    const float xn0 = x0 * sc * w[lane];
    const float xn1 = x1 * sc * w[lane + 32];
    const int p = posoff + row;
    const float c0 = cosp[(size_t)p * 64 + lane];
    const float c1 = cosp[(size_t)p * 64 + lane + 32];
    const float s0 = sinp[(size_t)p * 64 + lane];
    const float s1 = sinp[(size_t)p * 64 + lane + 32];
    _Float16* d = dst + ((size_t)h * rows + row) * 64;
    d[lane]      = (_Float16)(xn0 * c0 - xn1 * s0);
    d[lane + 32] = (_Float16)(xn1 * c1 + xn0 * s1);
}

// ---------------------------------------------------------------- V transpose
// vraw: (4096 x 512) f32 -> vt[kvh][d][t] f16  (d-major so PV B-frags are contiguous)
__global__ __launch_bounds__(256) void vtrans(const float* __restrict__ vraw,
                                              _Float16* __restrict__ vt) {
    const int i = blockIdx.x * 256 + threadIdx.x;
    if (i >= 8 * 64 * 4096) return;
    const int t   = i & 4095;
    const int d   = (i >> 12) & 63;
    const int kvh = i >> 18;
    vt[i] = (_Float16)vraw[(size_t)t * 512 + kvh * 64 + d];
}

// ---------------------------------------------------------------- Flash attention
// Block: 128 q-rows x 1 head. 8 waves; wave owns 16 q-rows x all keys.
__global__ __launch_bounds__(256) void flash_attn(const _Float16* __restrict__ Qf,
                                                  const _Float16* __restrict__ Kf,
                                                  const _Float16* __restrict__ Vt,
                                                  const float* __restrict__ mask,
                                                  const float* __restrict__ tb,
                                                  const int* __restrict__ rel,
                                                  _Float16* __restrict__ Of) {
    const int h     = blockIdx.y;
    const int q0    = blockIdx.x * 128;
    const int kvh   = h >> 1;
    const int wid   = threadIdx.x >> 5;
    const int lane  = threadIdx.x & 31;
    const int nl    = lane & 15;
    const int mb    = (lane >> 4) << 3;
    const int qrow0 = q0 + wid * 16;
    const int nb    = qrow0 >> 6;          // 64-block index of this wave's rows

    __shared__ _Float16 Ps[8 * 16 * 72];
    _Float16* ps = &Ps[wid * 16 * 72];

    const _Float16* Qh = Qf + (size_t)h * 2048 * 64;
    const _Float16* Kh = Kf + (size_t)kvh * 4096 * 64;
    const _Float16* Vh = Vt + (size_t)kvh * 64 * 4096;

    const v16h aq0 = load_frag(Qh + (size_t)qrow0 * 64, 64);
    const v16h aq1 = load_frag(Qh + (size_t)qrow0 * 64 + 32, 64);

    v8f o[4];
#pragma unroll
    for (int t = 0; t < 4; ++t) o[t] = zero8();
    float Mr[8], Sr[8];
#pragma unroll
    for (int r = 0; r < 8; ++r) { Mr[r] = -1e30f; Sr[r] = 0.f; }

    for (int kt = 0; kt < 64; ++kt) {
        const int k0 = kt * 64;
        v8f sc[4];
#pragma unroll
        for (int kk = 0; kk < 4; ++kk) sc[kk] = zero8();
        // ---- scores: Q(16x64) . K^T(64x64)
#pragma unroll
        for (int s = 0; s < 2; ++s) {
            const v16h a = s ? aq1 : aq0;
#pragma unroll
            for (int kk = 0; kk < 4; ++kk) {
                v16h bk = load_frag(Kh + (size_t)(k0 + kk * 16) * 64 + s * 32, 64);
                sc[kk] = wmma32(a, bk, sc[kk]);
            }
        }
        // ---- scale + mask + block-diagonal tree bias
        const bool hasb = (k0 >= 2048) && (((k0 - 2048) >> 6) == nb);
#pragma unroll
        for (int kk = 0; kk < 4; ++kk) {
            const int key = k0 + kk * 16 + nl;
#pragma unroll
            for (int r = 0; r < 8; ++r) {
                const int qrow = qrow0 + mb + r;
                float v = sc[kk][r] * 0.125f + mask[(size_t)qrow * 4096 + key];
                if (hasb) {
                    const int t1 = qrow & 63, t2 = (key - 2048) & 63;
                    v += tb[rel[(nb << 12) + (t1 << 6) + t2] * 16 + h];
                }
                sc[kk][r] = v;
            }
        }
        // ---- online softmax (rows live across 16 lanes of the same half-wave)
        float mnew[8];
#pragma unroll
        for (int r = 0; r < 8; ++r) {
            float mx = fmaxf(fmaxf(sc[0][r], sc[1][r]), fmaxf(sc[2][r], sc[3][r]));
#pragma unroll
            for (int off = 8; off >= 1; off >>= 1) mx = fmaxf(mx, __shfl_xor(mx, off, 32));
            mnew[r] = fmaxf(Mr[r], mx);
        }
#pragma unroll
        for (int r = 0; r < 8; ++r) {
            const float al = __expf(Mr[r] - mnew[r]);
            Mr[r] = mnew[r];
            Sr[r] *= al;
#pragma unroll
            for (int t = 0; t < 4; ++t) o[t][r] *= al;
        }
        float rs[8];
#pragma unroll
        for (int r = 0; r < 8; ++r) rs[r] = 0.f;
#pragma unroll
        for (int kk = 0; kk < 4; ++kk)
#pragma unroll
            for (int r = 0; r < 8; ++r) {
                const float p = __expf(sc[kk][r] - Mr[r]);
                sc[kk][r] = p;
                rs[r] += p;
            }
#pragma unroll
        for (int r = 0; r < 8; ++r) {
#pragma unroll
            for (int off = 8; off >= 1; off >>= 1) rs[r] += __shfl_xor(rs[r], off, 32);
            Sr[r] += rs[r];
        }
        // ---- C-layout -> A-layout via per-wave LDS stripe
#pragma unroll
        for (int kk = 0; kk < 4; ++kk)
#pragma unroll
            for (int r = 0; r < 8; ++r)
                ps[(mb + r) * 72 + kk * 16 + nl] = (_Float16)sc[kk][r];
        asm volatile("s_wait_dscnt 0" ::: "memory");
        // ---- O += P(16x64) . V(64x64)
#pragma unroll
        for (int s = 0; s < 2; ++s) {
            v16h ap = load_frag(ps + s * 32, 72);
#pragma unroll
            for (int d = 0; d < 4; ++d) {
                v16h bv = load_frag(Vh + (size_t)(d * 16) * 4096 + k0 + s * 32, 4096);
                o[d] = wmma32(ap, bv, o[d]);
            }
        }
    }
#pragma unroll
    for (int r = 0; r < 8; ++r) Sr[r] = 1.0f / Sr[r];
#pragma unroll
    for (int d = 0; d < 4; ++d)
#pragma unroll
        for (int r = 0; r < 8; ++r) {
            const int qrow = qrow0 + mb + r;
            Of[(size_t)qrow * 1024 + h * 64 + d * 16 + nl] = (_Float16)(o[d][r] * Sr[r]);
        }
}

// ---------------------------------------------------------------- host
extern "C" void kernel_launch(void* const* d_in, const int* in_sizes, int n_in,
                              void* d_out, int out_size, void* d_ws, size_t ws_size,
                              hipStream_t stream) {
    (void)in_sizes; (void)n_in; (void)out_size; (void)ws_size;
    const float* hidden = (const float*)d_in[0];
    const float* target = (const float*)d_in[1];
    const float* cosp   = (const float*)d_in[2];
    const float* sinp   = (const float*)d_in[3];
    const float* mask   = (const float*)d_in[4];
    const float* Wq     = (const float*)d_in[5];
    const float* Wk     = (const float*)d_in[6];
    const float* Wv     = (const float*)d_in[7];
    const float* Wo     = (const float*)d_in[8];
    const float* qw     = (const float*)d_in[9];
    const float* kw     = (const float*)d_in[10];
    const float* tb     = (const float*)d_in[11];
    const int*   rel    = (const int*)d_in[12];

    char* ws = (char*)d_ws;
    size_t off = 0;
    auto alloc = [&](size_t bytes) -> void* {
        void* p = ws + off;
        off += (bytes + 255) & ~(size_t)255;
        return p;
    };
    _Float16* Xf   = (_Float16*)alloc(4096ull * 1024 * 2);   // [target; hidden] f16
    _Float16* Wq16 = (_Float16*)alloc(1024ull * 1024 * 2);
    _Float16* Wk16 = (_Float16*)alloc(1024ull * 512 * 2);
    _Float16* Wv16 = (_Float16*)alloc(1024ull * 512 * 2);
    _Float16* Wo16 = (_Float16*)alloc(1024ull * 1024 * 2);
    float*    Qraw = (float*)alloc(2048ull * 1024 * 4);
    float*    Kraw = (float*)alloc(4096ull * 512 * 4);
    float*    Vraw = (float*)alloc(4096ull * 512 * 4);
    _Float16* Qf   = (_Float16*)alloc(16ull * 2048 * 64 * 2);
    _Float16* Kf2  = (_Float16*)alloc(8ull * 4096 * 64 * 2);
    _Float16* Vt   = (_Float16*)alloc(8ull * 64 * 4096 * 2);
    _Float16* Of   = (_Float16*)alloc(2048ull * 1024 * 2);

    auto cast = [&](const float* s, _Float16* d, int n) {
        cast_f16_kernel<<<(n + 255) / 256, 256, 0, stream>>>(s, d, n);
    };
    cast(target, Xf, 2048 * 1024);
    cast(hidden, Xf + 2048ull * 1024, 2048 * 1024);
    cast(Wq, Wq16, 1024 * 1024);
    cast(Wk, Wk16, 1024 * 512);
    cast(Wv, Wv16, 1024 * 512);
    cast(Wo, Wo16, 1024 * 1024);

    // projections
    gemm_wmma<<<dim3(16, 16), 256, 0, stream>>>(Xf + 2048ull * 1024, Wq16, Qraw, 2048, 1024, 1024);
    gemm_wmma<<<dim3(8, 32), 256, 0, stream>>>(Xf, Wk16, Kraw, 4096, 512, 1024);
    gemm_wmma<<<dim3(8, 32), 256, 0, stream>>>(Xf, Wv16, Vraw, 4096, 512, 1024);

    // RMSNorm + RoPE (q positions CTX..TOTAL-1; k positions 0..TOTAL-1)
    norm_rope<<<4096, 256, 0, stream>>>(Qraw, cosp, sinp, qw, Qf, 2048, 16, 1024, 2048);
    norm_rope<<<4096, 256, 0, stream>>>(Kraw, cosp, sinp, kw, Kf2, 4096, 8, 512, 0);
    vtrans<<<(8 * 64 * 4096) / 256, 256, 0, stream>>>(Vraw, Vt);

    // flash attention with tree bias
    flash_attn<<<dim3(16, 16), 256, 0, stream>>>(Qf, Kf2, Vt, mask, tb, rel, Of);

    // output projection
    gemm_wmma<<<dim3(16, 16), 256, 0, stream>>>(Of, Wo16, (float*)d_out, 2048, 1024, 1024);
}
